// CPCModel_36326833389992
// MI455X (gfx1250) — matrix operand order repaired and checked by reading, not compiled, not run on hardware
//
#include <hip/hip_runtime.h>
#include <math.h>

typedef __attribute__((ext_vector_type(16))) __bf16 v16bf;
typedef __attribute__((ext_vector_type(8)))  float  v8f;
typedef __attribute__((ext_vector_type(4)))  unsigned int u32x4;

#define GB0 2048
#define NTOT 4096

__device__ __forceinline__ unsigned short f32_to_bf16(float f) {
  unsigned int u = __float_as_uint(f);
  u += 0x7FFFu + ((u >> 16) & 1u);   // round-to-nearest-even
  return (unsigned short)(u >> 16);
}

union PunAB { struct { u32x4 a, b; } u; v16bf v; };

// A-operand (16x32 bf16 tile) from row-major [rows][128] bf16 matrix at (row0,k0).
// ISA layout: lanes 0-15 row M=lane, v0..v3 = K 0..7, v4..v7 = K16..23; upper half +8.
__device__ __forceinline__ v16bf load_tileA(const unsigned short* __restrict__ base,
                                            int row0, int k0) {
  int lane = threadIdx.x & 31;
  int r = row0 + (lane & 15);
  int hi8 = (lane >> 4) << 3;                 // 0 or 8
  const unsigned short* p = base + r * 128 + k0 + hi8;
  PunAB t;
  t.u.a = *reinterpret_cast<const u32x4*>(p);       // K = k0+hi8    .. +7
  t.u.b = *reinterpret_cast<const u32x4*>(p + 16);  // K = k0+16+hi8 .. +7
  return t.v;
}

// B-operand (32x16): B[k][n] = mat[n0+n][k], mat row-major [rows][128] bf16.
// ISA layout: lanes 0-15 hold K=0..7 and K=16..23, lanes 16-31 K=8..15 and K=24..31.
__device__ __forceinline__ v16bf load_tileB(const unsigned short* __restrict__ base,
                                            int n0, int k0) {
  int lane = threadIdx.x & 31;
  int n = n0 + (lane & 15);
  int off = k0 + ((lane >> 4) << 3);          // +8 for upper half
  const unsigned short* p = base + n * 128 + off;
  PunAB t;
  t.u.a = *reinterpret_cast<const u32x4*>(p);
  t.u.b = *reinterpret_cast<const u32x4*>(p + 16);
  return t.v;
}

__device__ __forceinline__ v8f wmma_bf16(v16bf a, v16bf b, v8f c) {
  return __builtin_amdgcn_wmma_f32_16x16x32_bf16(false, a, false, b,
                                                 (short)0, c, false, false);
}

// ---- pack zw (both groups) to bf16 [4096][128] -----------------------------
__global__ __launch_bounds__(256) void k_pack_zw(const float* __restrict__ zw0,
                                                 const float* __restrict__ zw1,
                                                 unsigned short* __restrict__ zwb) {
  int idx = blockIdx.x * 256 + threadIdx.x;
  if (idx >= NTOT * 128) return;
  int i = idx >> 7, k = idx & 127;
  float v = (i < GB0) ? zw0[i * 128 + k] : zw1[(i - GB0) * 128 + k];
  zwb[idx] = f32_to_bf16(v);
}

// ---- pack weights: linb[2][64][128], Bbuf[2][128][128]=[Ww|Wk], predbias ---
__global__ __launch_bounds__(256) void k_pack_w(
    const float* __restrict__ l0w, const float* __restrict__ l1w,
    const float* __restrict__ Ww0w, const float* __restrict__ Ww1w,
    const float* __restrict__ Wkw,
    const float* __restrict__ Ww0b, const float* __restrict__ Ww1b,
    const float* __restrict__ Wkb,
    unsigned short* __restrict__ linb, unsigned short* __restrict__ Bbuf,
    float* __restrict__ predbias) {
  int idx = blockIdx.x * 256 + threadIdx.x;
  if (idx < 2 * 64 * 128) {
    int g = idx >> 13;
    int rem = idx & 8191;
    const float* w = g ? l1w : l0w;             // (64,128) row-major
    linb[idx] = f32_to_bf16(w[rem]);
    return;
  }
  idx -= 2 * 64 * 128;
  if (idx < 2 * 128 * 128) {
    int g = idx >> 14;
    int rem = idx & 16383;
    int n = rem >> 7, k = rem & 127;
    float v;
    if (k < 64) { const float* Ww = g ? Ww1w : Ww0w; v = Ww[n * 64 + k]; }
    else        { v = Wkw[n * 64 + (k - 64)]; }
    Bbuf[(g << 14) + rem] = f32_to_bf16(v);
    return;
  }
  idx -= 2 * 128 * 128;
  if (idx < 256) {
    int g = idx >> 7, n = idx & 127;
    const float* Wb = g ? Ww1b : Ww0b;
    predbias[idx] = Wb[n] + Wkb[n];
  }
}

// ---- zt = relu(zw @ lin_w.T + b), WMMA bf16, out f32 [4096][64] ------------
__global__ __launch_bounds__(128) void k_gemm_zt(
    const unsigned short* __restrict__ zwb, const unsigned short* __restrict__ linb,
    const float* __restrict__ l0b, const float* __restrict__ l1b,
    float* __restrict__ zt) {
  int i0 = blockIdx.x * 16;
  int wave = threadIdx.x >> 5, lane = threadIdx.x & 31;
  int g = (i0 >= GB0) ? 1 : 0;
  const unsigned short* Bm = linb + g * 64 * 128;
  const float* lb = g ? l1b : l0b;
  int n0 = wave * 16;
  v8f acc = {0.f, 0.f, 0.f, 0.f, 0.f, 0.f, 0.f, 0.f};
#pragma unroll
  for (int kk = 0; kk < 4; ++kk)
    acc = wmma_bf16(load_tileA(zwb, i0, kk * 32), load_tileB(Bm, n0, kk * 32), acc);
  int half = lane >> 4;
  int n = n0 + (lane & 15);
  float bias = lb[n];
#pragma unroll
  for (int r = 0; r < 8; ++r) {
    float v = acc[r] + bias;
    zt[(i0 + r + 8 * half) * 64 + n] = v > 0.f ? v : 0.f;
  }
}

// ---- attention scores s[i] = tanh(zt_i @ a1w.T + a1b) @ a2w + a2b ----------
__global__ __launch_bounds__(256) void k_attn(
    const float* __restrict__ zt,
    const float* __restrict__ a01w, const float* __restrict__ a01b,
    const float* __restrict__ a02w, const float* __restrict__ a02b,
    const float* __restrict__ a11w, const float* __restrict__ a11b,
    const float* __restrict__ a12w, const float* __restrict__ a12b,
    float* __restrict__ sbuf) {
  int row = blockIdx.x * 8 + (threadIdx.x >> 5);
  int lane = threadIdx.x & 31;
  int g = row >= GB0;
  const float* a1w = g ? a11w : a01w;
  const float* a1b = g ? a11b : a01b;
  const float* a2w = g ? a12w : a02w;
  const float* a2b = g ? a12b : a02b;
  const float* z = zt + row * 64;
  float acc = a1b[lane];
  const float* w = a1w + lane * 64;
#pragma unroll
  for (int k = 0; k < 64; ++k) acc = fmaf(z[k], w[k], acc);
  float p = tanhf(acc) * a2w[lane];
#pragma unroll
  for (int m = 1; m < 32; m <<= 1) p += __shfl_xor(p, m, 32);
  if (lane == 0) sbuf[row] = p + a2b[0];
}

// ---- per-group stats: m=max(s), e_i=exp(s_i-m), S=sum e, T[64]=sum e*zt ----
__global__ __launch_bounds__(256) void k_gstats(
    const float* __restrict__ sbuf, const float* __restrict__ zt,
    float* __restrict__ ebuf, float* __restrict__ gstats) {
  int g = blockIdx.x;
  int t = threadIdx.x;
  __shared__ float red[256];
  const float* s = sbuf + g * GB0;
  float lm = -INFINITY;
  for (int i = t; i < GB0; i += 256) lm = fmaxf(lm, s[i]);
  red[t] = lm; __syncthreads();
  for (int o = 128; o > 0; o >>= 1) { if (t < o) red[t] = fmaxf(red[t], red[t + o]); __syncthreads(); }
  float m = red[0]; __syncthreads();
  float ls = 0.f;
  for (int i = t; i < GB0; i += 256) { float e = __expf(s[i] - m); ebuf[g * GB0 + i] = e; ls += e; }
  red[t] = ls; __syncthreads();
  for (int o = 128; o > 0; o >>= 1) { if (t < o) red[t] += red[t + o]; __syncthreads(); }
  float S = red[0]; __syncthreads();
  int k = t & 63, q = t >> 6;
  float Tp = 0.f;
  for (int i = q; i < GB0; i += 4) Tp += ebuf[g * GB0 + i] * zt[(g * GB0 + i) * 64 + k];
  red[t] = Tp; __syncthreads();
  if (t < 64) gstats[g * 66 + 2 + t] = red[t] + red[64 + t] + red[128 + t] + red[192 + t];
  if (t == 0) { gstats[g * 66 + 0] = m; gstats[g * 66 + 1] = S; }
}

// ---- Abuf[i] = bf16([pooled_i | c_i]);  pooled_i = (T - e_i*zt_i)/(S - e_i) -
__global__ __launch_bounds__(256) void k_abuf(
    const float* __restrict__ ebuf, const float* __restrict__ zt,
    const float* __restrict__ c, const float* __restrict__ gstats,
    unsigned short* __restrict__ Abuf) {
  int idx = blockIdx.x * 256 + threadIdx.x;   // 4096*64
  int i = idx >> 6, k = idx & 63;
  int g = i >= GB0;
  float e = ebuf[i];
  float S = gstats[g * 66 + 1];
  float T = gstats[g * 66 + 2 + k];
  float pooled = (T - e * zt[i * 64 + k]) / (S - e);
  Abuf[i * 128 + k] = f32_to_bf16(pooled);
  Abuf[i * 128 + 64 + k] = f32_to_bf16(c[i * 64 + k]);
}

// ---- pred = [pooled|c] @ [Ww;Wk].T + bias, out bf16 [4096][128] ------------
__global__ __launch_bounds__(256) void k_gemm_pred(
    const unsigned short* __restrict__ Abuf, const unsigned short* __restrict__ Bbuf,
    const float* __restrict__ predbias, unsigned short* __restrict__ predb) {
  int i0 = blockIdx.x * 16;
  int wave = threadIdx.x >> 5, lane = threadIdx.x & 31;
  int g = i0 >= GB0;
  const unsigned short* Bm = Bbuf + g * 128 * 128;
  int n0 = wave * 16;
  v8f acc = {0.f, 0.f, 0.f, 0.f, 0.f, 0.f, 0.f, 0.f};
#pragma unroll
  for (int kk = 0; kk < 4; ++kk)
    acc = wmma_bf16(load_tileA(Abuf, i0, kk * 32), load_tileB(Bm, n0, kk * 32), acc);
  int half = lane >> 4, n = n0 + (lane & 15);
  float bias = predbias[g * 128 + n];
#pragma unroll
  for (int r = 0; r < 8; ++r)
    predb[(i0 + r + 8 * half) * 128 + n] = f32_to_bf16(acc[r] + bias);
}

// ---- total = zw @ pred.T fused with row-wise logsumexp & diag --------------
// 2-way i-tile register blocking: 32 rows/block, each B tile feeds 8 WMMAs.
__global__ __launch_bounds__(256) void k_gemm_total(
    const unsigned short* __restrict__ zwb, const unsigned short* __restrict__ predb,
    float* __restrict__ partials) {
  int ib = blockIdx.x;                 // 0..127
  int i0 = ib * 32;
  int wave = threadIdx.x >> 5, lane = threadIdx.x & 31;
  int half = lane >> 4, col = lane & 15;
  __shared__ float lds_diag[32], lds_m[8 * 32], lds_s[8 * 32], lds_row[32];
  v16bf A[2][4];
#pragma unroll
  for (int ti = 0; ti < 2; ++ti)
#pragma unroll
    for (int kk = 0; kk < 4; ++kk)
      A[ti][kk] = load_tileA(zwb, i0 + ti * 16, kk * 32);
  float m[2][8], s[2][8];
#pragma unroll
  for (int ti = 0; ti < 2; ++ti)
#pragma unroll
    for (int r = 0; r < 8; ++r) { m[ti][r] = -INFINITY; s[ti][r] = 0.f; }

  for (int jt = wave; jt < 256; jt += 8) {
    int j0 = jt * 16;
    // Prefetch next j-tile's B rows into cache (global_prefetch_b8).
    int jn = jt + 8;
    if (jn < 256) __builtin_prefetch(predb + (jn * 16 + col) * 128, 0, 1);
    v16bf B0 = load_tileB(predb, j0, 0);
    v16bf B1 = load_tileB(predb, j0, 32);
    v16bf B2 = load_tileB(predb, j0, 64);
    v16bf B3 = load_tileB(predb, j0, 96);
#pragma unroll
    for (int ti = 0; ti < 2; ++ti) {
      v8f acc = {0.f, 0.f, 0.f, 0.f, 0.f, 0.f, 0.f, 0.f};
      acc = wmma_bf16(A[ti][0], B0, acc);
      acc = wmma_bf16(A[ti][1], B1, acc);
      acc = wmma_bf16(A[ti][2], B2, acc);
      acc = wmma_bf16(A[ti][3], B3, acc);
      bool diagTile = (jt == ib * 2 + ti);
#pragma unroll
      for (int r = 0; r < 8; ++r) {
        float x = acc[r];
        if (diagTile && col == r + 8 * half) lds_diag[ti * 16 + r + 8 * half] = x;
        if (x > m[ti][r]) {                       // online logsumexp, 1 exp/elem
          s[ti][r] = s[ti][r] * __expf(m[ti][r] - x) + 1.0f;
          m[ti][r] = x;
        } else {
          s[ti][r] += __expf(x - m[ti][r]);
        }
      }
    }
  }
  // merge (m,s) across the 16 lanes of each half (rows live within a half)
#pragma unroll
  for (int ti = 0; ti < 2; ++ti)
#pragma unroll
    for (int r = 0; r < 8; ++r)
#pragma unroll
      for (int step = 1; step < 16; step <<= 1) {
        float mo = __shfl_xor(m[ti][r], step, 32);
        float so = __shfl_xor(s[ti][r], step, 32);
        float nm = fmaxf(m[ti][r], mo);
        s[ti][r] = s[ti][r] * __expf(m[ti][r] - nm) + so * __expf(mo - nm);
        m[ti][r] = nm;
      }
  if (col == 0) {
#pragma unroll
    for (int ti = 0; ti < 2; ++ti)
#pragma unroll
      for (int r = 0; r < 8; ++r) {
        lds_m[wave * 32 + ti * 16 + r + 8 * half] = m[ti][r];
        lds_s[wave * 32 + ti * 16 + r + 8 * half] = s[ti][r];
      }
  }
  __syncthreads();
  int t = threadIdx.x;
  if (t < 32) {
    float M = -INFINITY, S = 0.f;
    for (int w = 0; w < 8; ++w) {
      float mw = lds_m[w * 32 + t], sw = lds_s[w * 32 + t];
      float nm = fmaxf(M, mw);
      S = S * __expf(M - nm) + sw * __expf(mw - nm);
      M = nm;
    }
    lds_row[t] = (M + __logf(S)) - lds_diag[t];   // lse - diag
  }
  __syncthreads();
  if (t == 0) {
    float a = 0.f;
    for (int q = 0; q < 32; ++q) a += lds_row[q];
    partials[ib] = a;
  }
}

__global__ __launch_bounds__(256) void k_final(const float* __restrict__ partials,
                                               float* __restrict__ out) {
  __shared__ float red[256];
  int t = threadIdx.x;
  red[t] = (t < 128) ? partials[t] : 0.f;
  __syncthreads();
  for (int o = 128; o > 0; o >>= 1) { if (t < o) red[t] += red[t + o]; __syncthreads(); }
  if (t == 0) out[0] = red[0] / 4096.0f;
}

extern "C" void kernel_launch(void* const* d_in, const int* in_sizes, int n_in,
                              void* d_out, int out_size, void* d_ws, size_t ws_size,
                              hipStream_t stream) {
  (void)in_sizes; (void)n_in; (void)out_size; (void)ws_size;
  const float* zw0  = (const float*)d_in[0];
  const float* zw1  = (const float*)d_in[1];
  const float* c    = (const float*)d_in[2];
  const float* Wkw  = (const float*)d_in[3];
  const float* Wkb  = (const float*)d_in[4];
  const float* Ww0w = (const float*)d_in[5];
  const float* Ww0b = (const float*)d_in[6];
  const float* Ww1w = (const float*)d_in[7];
  const float* Ww1b = (const float*)d_in[8];
  const float* l0w  = (const float*)d_in[9];
  const float* l0b  = (const float*)d_in[10];
  const float* l1w  = (const float*)d_in[11];
  const float* l1b  = (const float*)d_in[12];
  const float* a01w = (const float*)d_in[13];
  const float* a01b = (const float*)d_in[14];
  const float* a02w = (const float*)d_in[15];
  const float* a02b = (const float*)d_in[16];
  const float* a11w = (const float*)d_in[17];
  const float* a11b = (const float*)d_in[18];
  const float* a12w = (const float*)d_in[19];
  const float* a12b = (const float*)d_in[20];

  char* ws = (char*)d_ws;
  unsigned short* zwb   = (unsigned short*)(ws + 0);        // 4096*128 bf16
  unsigned short* Abuf  = (unsigned short*)(ws + 1048576);  // 4096*128 bf16
  unsigned short* predb = (unsigned short*)(ws + 2097152);  // 4096*128 bf16
  unsigned short* linb  = (unsigned short*)(ws + 3145728);  // 2*64*128 bf16
  unsigned short* Bbuf  = (unsigned short*)(ws + 3178496);  // 2*128*128 bf16
  float* predbias       = (float*)(ws + 3244032);           // 2*128 f32
  float* zt             = (float*)(ws + 3245056);           // 4096*64 f32
  float* sbuf           = (float*)(ws + 4293632);           // 4096 f32
  float* ebuf           = (float*)(ws + 4310016);           // 4096 f32
  float* gstats         = (float*)(ws + 4326400);           // 2*66 f32
  float* partials       = (float*)(ws + 4327424);           // 128 f32
  float* out = (float*)d_out;

  k_pack_zw<<<2048, 256, 0, stream>>>(zw0, zw1, zwb);
  k_pack_w<<<193, 256, 0, stream>>>(l0w, l1w, Ww0w, Ww1w, Wkw, Ww0b, Ww1b, Wkb,
                                    linb, Bbuf, predbias);
  k_gemm_zt<<<256, 128, 0, stream>>>(zwb, linb, l0b, l1b, zt);
  k_attn<<<512, 256, 0, stream>>>(zt, a01w, a01b, a02w, a02b,
                                  a11w, a11b, a12w, a12b, sbuf);
  k_gstats<<<2, 256, 0, stream>>>(sbuf, zt, ebuf, gstats);
  k_abuf<<<1024, 256, 0, stream>>>(ebuf, zt, c, gstats, Abuf);
  k_gemm_pred<<<256, 256, 0, stream>>>(Abuf, Bbuf, predbias, predb);
  k_gemm_total<<<128, 256, 0, stream>>>(zwb, predb, partials);
  k_final<<<1, 256, 0, stream>>>(partials, out);
}
